// VisionTransformer_22806276341828
// MI455X (gfx1250) — compile-verified, matrix operand order checked by reading, not run on hardware
//
#include <hip/hip_runtime.h>
#include <math.h>

// ---------- types ----------
typedef __bf16  v16bf __attribute__((ext_vector_type(16)));
typedef float   v8f   __attribute__((ext_vector_type(8)));

union BF16x16 { v16bf v; uint4 q[2]; unsigned short s[16]; };
union U32x    { uint4 q[4]; unsigned short s[32]; };

__device__ inline unsigned short f2bf(float f) {
    unsigned int u = __builtin_bit_cast(unsigned int, f);
    u += 0x7FFFu + ((u >> 16) & 1u);          // round-to-nearest-even
    return (unsigned short)(u >> 16);
}

__device__ inline v8f wmma_bf16(const BF16x16& a, const BF16x16& b, v8f c) {
    return __builtin_amdgcn_wmma_f32_16x16x32_bf16(false, a.v, false, b.v,
                                                   (short)0, c, false, false);
}

// ---------- model dims ----------
#define BATCH 16
#define NSEQ  577
#define NPATS 576
#define TOK   (BATCH * NSEQ)   // 9232
#define DD    768
#define D3    2304
#define DMLP  3072

// =====================================================================
// f32 -> bf16 bulk convert (4 elems / thread)
// =====================================================================
__global__ __launch_bounds__(256) void cvt_k(const float* __restrict__ in,
                                             unsigned short* __restrict__ out,
                                             long n4) {
    long idx = (long)blockIdx.x * 256 + threadIdx.x;
    if (idx >= n4) return;
    float4 v = reinterpret_cast<const float4*>(in)[idx];
    uint2 o;
    o.x = (unsigned)f2bf(v.x) | ((unsigned)f2bf(v.y) << 16);
    o.y = (unsigned)f2bf(v.z) | ((unsigned)f2bf(v.w) << 16);
    reinterpret_cast<uint2*>(out)[idx] = o;
}

// =====================================================================
// im2col: x[B,3,384,384] -> A[9216, 768] bf16  (kidx = c*256 + p*16 + q)
// =====================================================================
__global__ __launch_bounds__(256) void im2col_k(const float* __restrict__ x,
                                                unsigned short* __restrict__ out) {
    int idx = blockIdx.x * 256 + threadIdx.x;
    if (idx >= 9216 * 768) return;
    int tokn = idx / 768, kidx = idx % 768;
    int b = tokn / 576, pr = tokn % 576;
    int gh = pr / 24, gw = pr % 24;
    int c = kidx / 256, rem = kidx % 256;
    int p = rem / 16, q = rem % 16;
    float v = x[(((size_t)b * 3 + c) * 384 + gh * 16 + p) * 384 + gw * 16 + q];
    out[idx] = f2bf(v);
}

// =====================================================================
// assemble h = concat(cls, tok) + pos  -> f32 [9232,768]
// =====================================================================
__global__ __launch_bounds__(256) void assemble_k(const unsigned short* __restrict__ tok,
                                                  const float* __restrict__ cls,
                                                  const float* __restrict__ pos,
                                                  float* __restrict__ h) {
    int idx = blockIdx.x * 256 + threadIdx.x;
    if (idx >= TOK * DD) return;
    int d = idx % DD;
    int tk = idx / DD;
    int b = tk / NSEQ, n = tk % NSEQ;
    float v;
    if (n == 0) v = cls[d];
    else {
        unsigned short us = tok[((size_t)(b * NPATS + n - 1)) * DD + d];
        v = __builtin_bit_cast(float, (unsigned)us << 16);
    }
    h[idx] = v + pos[(size_t)n * DD + d];
}

// =====================================================================
// LayerNorm: f32 in, bf16 (OUTBF=1) or f32 (OUTBF=0) out. 1 row / block.
// =====================================================================
template <int OUTBF>
__global__ __launch_bounds__(128) void ln_k(const float* __restrict__ x,
                                            const float* __restrict__ g,
                                            const float* __restrict__ bta,
                                            unsigned short* outb, float* outf) {
    __shared__ float s1[128], s2[128];
    int row = blockIdx.x, t = threadIdx.x;
    const float* xr = x + (size_t)row * DD;
    float vals[6], su = 0.f, sq = 0.f;
#pragma unroll
    for (int i = 0; i < 6; ++i) {
        float v = xr[t + i * 128];
        vals[i] = v; su += v; sq += v * v;
    }
    s1[t] = su; s2[t] = sq; __syncthreads();
    for (int off = 64; off > 0; off >>= 1) {
        if (t < off) { s1[t] += s1[t + off]; s2[t] += s2[t + off]; }
        __syncthreads();
    }
    float mu = s1[0] * (1.0f / 768.0f);
    float var = s2[0] * (1.0f / 768.0f) - mu * mu;
    float rs = rsqrtf(var + 1e-6f);
#pragma unroll
    for (int i = 0; i < 6; ++i) {
        int c = t + i * 128;
        float y = (vals[i] - mu) * rs * g[c] + bta[c];
        if (OUTBF) outb[(size_t)row * DD + c] = f2bf(y);
        else       outf[(size_t)row * DD + c] = y;
    }
}

// =====================================================================
// GEMM: out[M,Nn] = A(bf16,[M,K]) * W([Nn,K])^T + bias
//   WBF=1: W is bf16; WBF=0: W is f32 (converted during staging)
//   EPI 0: bf16 store  1: bf16 + exact GELU  2: f32 residual add
// 256 thr (8 waves), block tile 128x128, BK=64, LDS ping-pong,
// wave tile 32x64 -> 16 WMMAs per K-step between single barriers.
// =====================================================================
#define LSTR 72   // 64 + 8 pad (halves); rows 144B -> 16B aligned

__device__ inline void stage_a(const unsigned short* __restrict__ A, int grow,
                               int K, int k0, int coff, int M, uint4 r[4]) {
    uint4 z = {0, 0, 0, 0};
    if (grow < M) {
        const uint4* p = reinterpret_cast<const uint4*>(A + (size_t)grow * K + k0 + coff);
#pragma unroll
        for (int i = 0; i < 4; ++i) r[i] = p[i];
    } else {
#pragma unroll
        for (int i = 0; i < 4; ++i) r[i] = z;
    }
}

__device__ inline void stage_w_bf(const unsigned short* __restrict__ W, int gn,
                                  int K, int k0, int coff, uint4 r[4]) {
    const uint4* p = reinterpret_cast<const uint4*>(W + (size_t)gn * K + k0 + coff);
#pragma unroll
    for (int i = 0; i < 4; ++i) r[i] = p[i];
}

__device__ inline void stage_w_f32(const float* __restrict__ W, int gn,
                                   int K, int k0, int coff, uint4 r[4]) {
    const float* p = W + (size_t)gn * K + k0 + coff;
    unsigned short tmp[32];
#pragma unroll
    for (int i = 0; i < 32; ++i) tmp[i] = f2bf(p[i]);
#pragma unroll
    for (int i = 0; i < 4; ++i) r[i] = reinterpret_cast<const uint4*>(tmp)[i];
}

template <int EPI, int WBF>
__global__ __launch_bounds__(256) void gemm_k(const unsigned short* __restrict__ A,
                                              const void* __restrict__ Wv,
                                              const float* __restrict__ bias,
                                              const float* resid,
                                              unsigned short* outb, float* outf,
                                              int M, int Nn, int K) {
    __shared__ __align__(16) unsigned short As[2][128 * LSTR];
    __shared__ __align__(16) unsigned short Ws[2][128 * LSTR];
    const int t = threadIdx.x;
    const int lane = t & 31, wv = t >> 5;
    const int wm = wv >> 1, wn = wv & 1;
    const int l15 = lane & 15, lh = lane >> 4;
    const int rbase = blockIdx.y * 128;
    const int nbase = blockIdx.x * 128;
    const int srow = t >> 1, scoff = (t & 1) * 32;   // 32 halves per thread
    const int grow = rbase + srow;
    const int gn = nbase + srow;

    const unsigned short* Wb = (const unsigned short*)Wv;
    const float* Wf = (const float*)Wv;

    v8f c[2][4];
    const v8f zc = {0.f, 0.f, 0.f, 0.f, 0.f, 0.f, 0.f, 0.f};
#pragma unroll
    for (int i = 0; i < 2; ++i)
#pragma unroll
        for (int j = 0; j < 4; ++j) c[i][j] = zc;

    uint4 ra[4], rw[4];
    // ---- preload tile 0 ----
    stage_a(A, grow, K, 0, scoff, M, ra);
    if (WBF) stage_w_bf(Wb, gn, K, 0, scoff, rw);
    else     stage_w_f32(Wf, gn, K, 0, scoff, rw);
#pragma unroll
    for (int i = 0; i < 4; ++i) {
        *reinterpret_cast<uint4*>(&As[0][srow * LSTR + scoff + i * 8]) = ra[i];
        *reinterpret_cast<uint4*>(&Ws[0][srow * LSTR + scoff + i * 8]) = rw[i];
    }
    __syncthreads();

    int p = 0;
    for (int k0 = 0; k0 < K; k0 += 64) {
        const bool more = (k0 + 64) < K;
        if (more) {
            // prefetch tile after next into GL2 (speculative)
            if (k0 + 128 < K) {
                __builtin_prefetch(A + (size_t)grow * K + k0 + 128 + scoff, 0, 1);
                if (WBF) __builtin_prefetch(Wb + (size_t)gn * K + k0 + 128 + scoff, 0, 1);
                else     __builtin_prefetch(Wf + (size_t)gn * K + k0 + 128 + scoff, 0, 1);
            }
            stage_a(A, grow, K, k0 + 64, scoff, M, ra);
            if (WBF) stage_w_bf(Wb, gn, K, k0 + 64, scoff, rw);
            else     stage_w_f32(Wf, gn, K, k0 + 64, scoff, rw);
#pragma unroll
            for (int i = 0; i < 4; ++i) {
                *reinterpret_cast<uint4*>(&As[1 - p][srow * LSTR + scoff + i * 8]) = ra[i];
                *reinterpret_cast<uint4*>(&Ws[1 - p][srow * LSTR + scoff + i * 8]) = rw[i];
            }
        }
        // ---- compute on buffer p (2 K-steps of 32) ----
#pragma unroll
        for (int ks = 0; ks < 2; ++ks) {
            BF16x16 af[2], wf[4];
#pragma unroll
            for (int tm = 0; tm < 2; ++tm) {
                int m = wm * 32 + tm * 16 + l15;
                af[tm].q[0] = *reinterpret_cast<const uint4*>(&As[p][m * LSTR + ks * 32 + lh * 8]);
                af[tm].q[1] = *reinterpret_cast<const uint4*>(&As[p][m * LSTR + ks * 32 + lh * 8 + 16]);
            }
#pragma unroll
            for (int tn = 0; tn < 4; ++tn) {
                int n = wn * 64 + tn * 16 + l15;
                wf[tn].q[0] = *reinterpret_cast<const uint4*>(&Ws[p][n * LSTR + ks * 32 + lh * 8]);
                wf[tn].q[1] = *reinterpret_cast<const uint4*>(&Ws[p][n * LSTR + ks * 32 + lh * 8 + 16]);
            }
#pragma unroll
            for (int tm = 0; tm < 2; ++tm)
#pragma unroll
                for (int tn = 0; tn < 4; ++tn)
                    c[tm][tn] = wmma_bf16(af[tm], wf[tn], c[tm][tn]);
        }
        __syncthreads();
        p ^= 1;
    }

    // ---- epilogue ----
#pragma unroll
    for (int tm = 0; tm < 2; ++tm) {
#pragma unroll
        for (int tn = 0; tn < 4; ++tn) {
            int gcol = nbase + wn * 64 + tn * 16 + l15;
            float bv = bias[gcol];
#pragma unroll
            for (int r = 0; r < 8; ++r) {
                int orow = rbase + wm * 32 + tm * 16 + lh * 8 + r;
                if (orow < M) {
                    float v = c[tm][tn][r] + bv;
                    if (EPI == 1) v = 0.5f * v * (1.0f + erff(v * 0.70710678118654752f));
                    if (EPI == 2)
                        outf[(size_t)orow * Nn + gcol] = resid[(size_t)orow * Nn + gcol] + v;
                    else
                        outb[(size_t)orow * Nn + gcol] = f2bf(v);
                }
            }
        }
    }
}

// =====================================================================
// Flash attention: grid (qtile=10, h=12, b=16), 128 thr (4 waves x 16 q-rows)
// =====================================================================
__global__ __launch_bounds__(128) void attn_k(const unsigned short* __restrict__ qkv,
                                              unsigned short* __restrict__ out) {
    __shared__ __align__(16) unsigned short Kc[64 * 72];
    __shared__ __align__(16) unsigned short Vt[64 * 72];
    __shared__ __align__(16) unsigned short Ps[4 * 16 * 72];
    const int t = threadIdx.x, lane = t & 31, w = t >> 5;
    const int l15 = lane & 15, lh = lane >> 4;
    const int qt = blockIdx.x, hh = blockIdx.y, b = blockIdx.z;
    const size_t base = (size_t)b * NSEQ * D3;

    BF16x16 aq[2];
    {
        int qn = qt * 64 + w * 16 + l15;
        if (qn < NSEQ) {
            const unsigned short* qp = qkv + base + (size_t)qn * D3 + hh * 64;
#pragma unroll
            for (int kk = 0; kk < 2; ++kk) {
                aq[kk].q[0] = *reinterpret_cast<const uint4*>(qp + kk * 32 + lh * 8);
                aq[kk].q[1] = *reinterpret_cast<const uint4*>(qp + kk * 32 + lh * 8 + 16);
            }
        } else {
            uint4 z = {0, 0, 0, 0};
            aq[0].q[0] = z; aq[0].q[1] = z; aq[1].q[0] = z; aq[1].q[1] = z;
        }
    }

    const v8f zc = {0.f, 0.f, 0.f, 0.f, 0.f, 0.f, 0.f, 0.f};
    float mrow[8], lrow[8];
    v8f oacc[4];
#pragma unroll
    for (int r = 0; r < 8; ++r) { mrow[r] = -1e30f; lrow[r] = 0.f; }
#pragma unroll
    for (int tn = 0; tn < 4; ++tn) oacc[tn] = zc;

    const int skr = t >> 1, sc0 = (t & 1) * 32;

    for (int j = 0; j < 10; ++j) {
        const int kb0 = j * 64;
        {
            int keyn = kb0 + skr;
            uint4 kd[4]; U32x vu;
            if (keyn < NSEQ) {
                const unsigned short* kp = qkv + base + (size_t)keyn * D3 + 768 + hh * 64 + sc0;
                const unsigned short* vp = qkv + base + (size_t)keyn * D3 + 1536 + hh * 64 + sc0;
#pragma unroll
                for (int i = 0; i < 4; ++i) {
                    kd[i]   = reinterpret_cast<const uint4*>(kp)[i];
                    vu.q[i] = reinterpret_cast<const uint4*>(vp)[i];
                }
            } else {
                uint4 z = {0, 0, 0, 0};
#pragma unroll
                for (int i = 0; i < 4; ++i) { kd[i] = z; vu.q[i] = z; }
            }
#pragma unroll
            for (int i = 0; i < 4; ++i)
                *reinterpret_cast<uint4*>(&Kc[skr * 72 + sc0 + i * 8]) = kd[i];
#pragma unroll
            for (int i = 0; i < 32; ++i) Vt[(sc0 + i) * 72 + skr] = vu.s[i];
        }
        __syncthreads();

        v8f s[4];
#pragma unroll
        for (int tn = 0; tn < 4; ++tn) s[tn] = zc;
#pragma unroll
        for (int tn = 0; tn < 4; ++tn) {
            int kl = tn * 16 + l15;
#pragma unroll
            for (int kk = 0; kk < 2; ++kk) {
                BF16x16 kf;
                kf.q[0] = *reinterpret_cast<const uint4*>(&Kc[kl * 72 + kk * 32 + lh * 8]);
                kf.q[1] = *reinterpret_cast<const uint4*>(&Kc[kl * 72 + kk * 32 + lh * 8 + 16]);
                s[tn] = wmma_bf16(aq[kk], kf, s[tn]);
            }
        }
#pragma unroll
        for (int tn = 0; tn < 4; ++tn) {
            bool msk = (kb0 + tn * 16 + l15) >= NSEQ;
#pragma unroll
            for (int r = 0; r < 8; ++r) {
                float v = s[tn][r] * 0.125f;
                s[tn][r] = msk ? -1e30f : v;
            }
        }
#pragma unroll
        for (int r = 0; r < 8; ++r) {
            float mx = fmaxf(fmaxf(s[0][r], s[1][r]), fmaxf(s[2][r], s[3][r]));
#pragma unroll
            for (int off = 1; off < 16; off <<= 1) mx = fmaxf(mx, __shfl_xor(mx, off, 32));
            float mnew = fmaxf(mrow[r], mx);
            float fs = __expf(mrow[r] - mnew);
            mrow[r] = mnew;
            float sum = 0.f;
#pragma unroll
            for (int tn = 0; tn < 4; ++tn) {
                float pv = __expf(s[tn][r] - mnew);
                s[tn][r] = pv; sum += pv;
            }
#pragma unroll
            for (int off = 1; off < 16; off <<= 1) sum += __shfl_xor(sum, off, 32);
            lrow[r] = lrow[r] * fs + sum;
#pragma unroll
            for (int tn = 0; tn < 4; ++tn) oacc[tn][r] *= fs;
        }
        unsigned short* myP = &Ps[w * 16 * 72];
#pragma unroll
        for (int tn = 0; tn < 4; ++tn)
#pragma unroll
            for (int r = 0; r < 8; ++r)
                myP[(lh * 8 + r) * 72 + tn * 16 + l15] = f2bf(s[tn][r]);
#pragma unroll
        for (int tn = 0; tn < 4; ++tn) {
            int nh = tn * 16 + l15;
#pragma unroll
            for (int kk = 0; kk < 2; ++kk) {
                BF16x16 pf, vf;
                pf.q[0] = *reinterpret_cast<const uint4*>(&myP[l15 * 72 + kk * 32 + lh * 8]);
                pf.q[1] = *reinterpret_cast<const uint4*>(&myP[l15 * 72 + kk * 32 + lh * 8 + 16]);
                vf.q[0] = *reinterpret_cast<const uint4*>(&Vt[nh * 72 + kk * 32 + lh * 8]);
                vf.q[1] = *reinterpret_cast<const uint4*>(&Vt[nh * 72 + kk * 32 + lh * 8 + 16]);
                oacc[tn] = wmma_bf16(pf, vf, oacc[tn]);
            }
        }
        __syncthreads();
    }
#pragma unroll
    for (int r = 0; r < 8; ++r) {
        int qn = qt * 64 + w * 16 + lh * 8 + r;
        if (qn < NSEQ) {
            float inv = 1.0f / lrow[r];
            size_t rowo = ((size_t)b * NSEQ + qn) * DD + hh * 64;
#pragma unroll
            for (int tn = 0; tn < 4; ++tn)
                out[rowo + tn * 16 + l15] = f2bf(oacc[tn][r] * inv);
        }
    }
}

// =====================================================================
// host-side launch
// =====================================================================
extern "C" void kernel_launch(void* const* d_in, const int* in_sizes, int n_in,
                              void* d_out, int out_size, void* d_ws, size_t ws_size,
                              hipStream_t stream) {
    const float* x      = (const float*)d_in[0];
    const float* conv_w = (const float*)d_in[1];
    const float* conv_b = (const float*)d_in[2];
    const float* cls    = (const float*)d_in[3];
    const float* pos    = (const float*)d_in[4];
    const float* ln1_g  = (const float*)d_in[5];
    const float* ln1_b  = (const float*)d_in[6];
    const float* qkv_w  = (const float*)d_in[7];
    const float* qkv_b  = (const float*)d_in[8];
    const float* proj_w = (const float*)d_in[9];
    const float* proj_b = (const float*)d_in[10];
    const float* ln2_g  = (const float*)d_in[11];
    const float* ln2_b  = (const float*)d_in[12];
    const float* fc1_w  = (const float*)d_in[13];
    const float* fc1_b  = (const float*)d_in[14];
    const float* fc2_w  = (const float*)d_in[15];
    const float* fc2_b  = (const float*)d_in[16];
    const float* lnf_g  = (const float*)d_in[17];
    const float* lnf_b  = (const float*)d_in[18];

    char* p0 = (char*)d_ws;
    char* p = p0;
    auto carve = [&](size_t bytes) {
        void* r = (void*)p;
        p += (bytes + 255) & ~(size_t)255;
        return r;
    };
    unsigned short* im2c = (unsigned short*)carve((size_t)9216 * DD * 2);
    unsigned short* tok  = (unsigned short*)carve((size_t)9216 * DD * 2);
    float*          h    = (float*)carve((size_t)TOK * DD * 4);
    unsigned short* y    = (unsigned short*)carve((size_t)TOK * DD * 2);
    unsigned short* qkvb = (unsigned short*)carve((size_t)TOK * D3 * 2);
    unsigned short* att  = (unsigned short*)carve((size_t)TOK * DD * 2);
    unsigned short* mlp  = (unsigned short*)carve((size_t)TOK * DMLP * 2);

    // bf16 weight mirrors (if workspace allows)
    const size_t n_conv = (size_t)DD * 768;
    const size_t n_qkv  = (size_t)12 * D3 * DD;
    const size_t n_proj = (size_t)12 * DD * DD;
    const size_t n_fc1  = (size_t)12 * DMLP * DD;
    const size_t n_fc2  = (size_t)12 * DD * DMLP;
    unsigned short* convb = (unsigned short*)carve(n_conv * 2);
    unsigned short* qkvwb = (unsigned short*)carve(n_qkv * 2);
    unsigned short* projb = (unsigned short*)carve(n_proj * 2);
    unsigned short* fc1b  = (unsigned short*)carve(n_fc1 * 2);
    unsigned short* fc2b  = (unsigned short*)carve(n_fc2 * 2);
    const bool wbf = ((size_t)(p - p0) <= ws_size);

    const int MT = (TOK + 127) / 128;   // 73

    if (wbf) {
        auto cvt = [&](const float* src, unsigned short* dst, size_t n) {
            long n4 = (long)(n / 4);
            cvt_k<<<(n4 + 255) / 256, 256, 0, stream>>>(src, dst, n4);
        };
        cvt(conv_w, convb, n_conv);
        cvt(qkv_w, qkvwb, n_qkv);
        cvt(proj_w, projb, n_proj);
        cvt(fc1_w, fc1b, n_fc1);
        cvt(fc2_w, fc2b, n_fc2);
    }

    im2col_k<<<(9216 * 768 + 255) / 256, 256, 0, stream>>>(x, im2c);
    if (wbf)
        gemm_k<0, 1><<<dim3(DD / 128, 9216 / 128), 256, 0, stream>>>(
            im2c, convb, conv_b, nullptr, tok, nullptr, 9216, DD, 768);
    else
        gemm_k<0, 0><<<dim3(DD / 128, 9216 / 128), 256, 0, stream>>>(
            im2c, conv_w, conv_b, nullptr, tok, nullptr, 9216, DD, 768);
    assemble_k<<<(TOK * DD + 255) / 256, 256, 0, stream>>>(tok, cls, pos, h);

    for (int l = 0; l < 12; ++l) {
        const float* g1 = ln1_g + (size_t)l * DD;
        const float* b1 = ln1_b + (size_t)l * DD;
        const float* bq = qkv_b + (size_t)l * D3;
        const float* bp = proj_b + (size_t)l * DD;
        const float* g2 = ln2_g + (size_t)l * DD;
        const float* b2 = ln2_b + (size_t)l * DD;
        const float* bb1 = fc1_b + (size_t)l * DMLP;
        const float* bb2 = fc2_b + (size_t)l * DD;

        ln_k<1><<<TOK, 128, 0, stream>>>(h, g1, b1, y, nullptr);
        if (wbf)
            gemm_k<0, 1><<<dim3(D3 / 128, MT), 256, 0, stream>>>(
                y, qkvwb + (size_t)l * D3 * DD, bq, nullptr, qkvb, nullptr, TOK, D3, DD);
        else
            gemm_k<0, 0><<<dim3(D3 / 128, MT), 256, 0, stream>>>(
                y, qkv_w + (size_t)l * D3 * DD, bq, nullptr, qkvb, nullptr, TOK, D3, DD);

        attn_k<<<dim3(10, 12, 16), 128, 0, stream>>>(qkvb, att);

        if (wbf)
            gemm_k<2, 1><<<dim3(DD / 128, MT), 256, 0, stream>>>(
                att, projb + (size_t)l * DD * DD, bp, h, nullptr, h, TOK, DD, DD);
        else
            gemm_k<2, 0><<<dim3(DD / 128, MT), 256, 0, stream>>>(
                att, proj_w + (size_t)l * DD * DD, bp, h, nullptr, h, TOK, DD, DD);

        ln_k<1><<<TOK, 128, 0, stream>>>(h, g2, b2, y, nullptr);

        if (wbf)
            gemm_k<1, 1><<<dim3(DMLP / 128, MT), 256, 0, stream>>>(
                y, fc1b + (size_t)l * DMLP * DD, bb1, nullptr, mlp, nullptr, TOK, DMLP, DD);
        else
            gemm_k<1, 0><<<dim3(DMLP / 128, MT), 256, 0, stream>>>(
                y, fc1_w + (size_t)l * DMLP * DD, bb1, nullptr, mlp, nullptr, TOK, DMLP, DD);

        if (wbf)
            gemm_k<2, 1><<<dim3(DD / 128, MT), 256, 0, stream>>>(
                mlp, fc2b + (size_t)l * DD * DMLP, bb2, h, nullptr, h, TOK, DD, DMLP);
        else
            gemm_k<2, 0><<<dim3(DD / 128, MT), 256, 0, stream>>>(
                mlp, fc2_w + (size_t)l * DD * DMLP, bb2, h, nullptr, h, TOK, DD, DMLP);
    }
    ln_k<0><<<TOK, 128, 0, stream>>>(h, lnf_g, lnf_b, nullptr, (float*)d_out);
}
